// EPA_65635690217518
// MI455X (gfx1250) — matrix operand order repaired and checked by reading, not compiled, not run on hardware
//
#include <hip/hip_runtime.h>

#define B_    2
#define N_    2048
#define C_    1024
#define H_    16
#define D_    64
#define R_    21
#define MTOK  (B_ * N_)   // 4096

typedef __attribute__((ext_vector_type(16))) __bf16 v16bf;
typedef __attribute__((ext_vector_type(8)))  float  v8f;
typedef __attribute__((ext_vector_type(4)))  unsigned int u32x4;
typedef __attribute__((ext_vector_type(8)))  int        i32x8;
typedef __attribute__((ext_vector_type(4)))  int        i32x4;

#if __has_builtin(__builtin_amdgcn_tensor_load_to_lds) && \
    __has_builtin(__builtin_amdgcn_s_wait_tensorcnt)
#define HAVE_TDM 1
#else
#define HAVE_TDM 0
#endif

union Frag16 { v16bf v; uint4 q[2]; };

__device__ __forceinline__ unsigned short f2bf(float f) {
    unsigned int u = __float_as_uint(f);
    u += 0x7FFFu + ((u >> 16) & 1u);           // round-to-nearest-even
    return (unsigned short)(u >> 16);
}
__device__ __forceinline__ float bf2f(unsigned short h) {
    return __uint_as_float(((unsigned int)h) << 16);
}

__device__ __forceinline__ v8f wmma_bf16(const Frag16& a, const Frag16& b, v8f c) {
    return __builtin_amdgcn_wmma_f32_16x16x32_bf16(false, a.v, false, b.v,
                                                   (short)0, c, false, false);
}

// LDS aperture maps flat addr[31:0] directly to the wave-relative LDS byte
// address (ISA 10.2 aperture table), so truncation yields the D# lds_addr.
__device__ __forceinline__ unsigned lds_off32(const void* p) {
    return (unsigned)(unsigned long long)p;
}

#if HAVE_TDM
// Issue a 2D TDM tile load: tile_d1 rows x tile_d0 elements (2-byte elems),
// row pitch stride_d0 elements, into LDS at lds_addr. Wave-uniform args.
// This toolchain's builtin arity: (u32x4 g0, i32x8 g1, i32x4, i32x4, i32x8, i32 cpol)
__device__ __forceinline__ void tdm_load_2d(unsigned lds_addr, const void* gptr,
                                            unsigned tensor_d0, unsigned tensor_d1,
                                            unsigned tile_d0, unsigned tile_d1,
                                            unsigned stride_d0) {
    unsigned long long ga = (unsigned long long)gptr;
    u32x4 g0;
    g0[0] = 1u;                                  // count=1, is_restore=0, no gather
    g0[1] = lds_addr;                            // LDS byte address
    g0[2] = (unsigned)ga;                        // global_addr[31:0]
    g0[3] = (unsigned)((ga >> 32) & 0x01FFFFFFu) // global_addr[56:32]
            | 0x80000000u;                       // type=2 ("image")
    i32x8 g1;
    g1[0] = (int)(1u << 16);                     // data_size=1 -> 2 bytes
    g1[1] = (int)((tensor_d0 & 0xFFFFu) << 16);  // tensor_dim0[15:0]
    g1[2] = (int)((tensor_d0 >> 16) | ((tensor_d1 & 0xFFFFu) << 16));
    g1[3] = (int)((tensor_d1 >> 16) | (tile_d0 << 16));
    g1[4] = (int)(tile_d1 & 0xFFFFu);            // tile_dim1 ; tile_dim2=0
    g1[5] = (int)stride_d0;                      // tensor_dim0_stride[31:0]
    g1[6] = 0;                                   // stride hi / dim1_stride lo
    g1[7] = 0;
    i32x4 z4; z4[0] = 0; z4[1] = 0; z4[2] = 0; z4[3] = 0;
    i32x8 z8;
    #pragma unroll
    for (int i = 0; i < 8; ++i) z8[i] = 0;
    __builtin_amdgcn_tensor_load_to_lds(g0, g1, z4, z4, z8, 0);
}
#endif

// ---------------------------------------------------------------- converts
__global__ void k_f32_to_bf16(const float* __restrict__ src,
                              unsigned short* __restrict__ dst, int n) {
    int i = blockIdx.x * blockDim.x + threadIdx.x;
    int st = gridDim.x * blockDim.x;
    for (; i < n; i += st) dst[i] = f2bf(src[i]);
}

__global__ void k_halfsum_bf16(const unsigned short* __restrict__ a,
                               const unsigned short* __restrict__ b,
                               unsigned short* __restrict__ o, int n) {
    int i = blockIdx.x * blockDim.x + threadIdx.x;
    int st = gridDim.x * blockDim.x;
    for (; i < n; i += st) o[i] = f2bf(0.5f * (bf2f(a[i]) + bf2f(b[i])));
}

// ---------------------------------------------------------------- WMMA GEMM
// C[M,Nn] = A[M,K] * B[K,Nn] + bias ; A,B bf16 row-major; M%128==0, Nn%128==0, K%32==0
__global__ void __launch_bounds__(128)
k_gemm_bf16(const unsigned short* __restrict__ A,
            const unsigned short* __restrict__ Bm,
            const float* __restrict__ bias,
            float* __restrict__ Cf, unsigned short* __restrict__ Cb,
            int M, int Nn, int K)
{
    __shared__ __align__(16) unsigned short As[128 * 32];  // [m_local][k_local]
    __shared__ __align__(16) unsigned short Bs[128 * 32];  // [n_local][k_local] (transposed)

    const int t = threadIdx.x;
    const int w = t >> 5, lane = t & 31;
    const int wr = w >> 1, wc = w & 1;          // wave quadrant in 128x128
    const int lm = lane & 15, hi = lane >> 4;
    const int m0 = blockIdx.x * 128;
    const int n0 = blockIdx.y * 128;

    v8f acc[4][4];
    #pragma unroll
    for (int i = 0; i < 4; ++i)
        #pragma unroll
        for (int j = 0; j < 4; ++j)
            #pragma unroll
            for (int e = 0; e < 8; ++e) acc[i][j][e] = 0.f;

    for (int k0 = 0; k0 < K; k0 += 32) {
#if HAVE_TDM
        // A tile DMA: 128 rows x 32 halves, row pitch K -- issued by wave 0 only
        if (w == 0) {
            tdm_load_2d(lds_off32(As), A + (size_t)m0 * K + k0,
                        (unsigned)K, (unsigned)M, 32u, 128u, (unsigned)K);
        }
#else
        {
            const uint4* s = (const uint4*)(A + (size_t)(m0 + t) * K + k0);
            uint4* d = (uint4*)&As[t * 32];
            d[0] = s[0]; d[1] = s[1]; d[2] = s[2]; d[3] = s[3];
        }
#endif
        // stage B tile transposed: 32 rows x 128 cols -> Bs[n][k]
        #pragma unroll
        for (int i = 0; i < 8; ++i) {
            int c  = t + 128 * i;                 // 0..1023
            int kr = c >> 5;                      // k-row 0..31
            int cc = c & 31;                      // col chunk (4 halves)
            uint2 dv = *(const uint2*)(Bm + (size_t)(k0 + kr) * Nn + n0 + 4 * cc);
            int nb = 4 * cc;
            Bs[(nb + 0) * 32 + kr] = (unsigned short)(dv.x & 0xFFFF);
            Bs[(nb + 1) * 32 + kr] = (unsigned short)(dv.x >> 16);
            Bs[(nb + 2) * 32 + kr] = (unsigned short)(dv.y & 0xFFFF);
            Bs[(nb + 3) * 32 + kr] = (unsigned short)(dv.y >> 16);
        }
#if HAVE_TDM
        if (w == 0) __builtin_amdgcn_s_wait_tensorcnt(0);
#endif
        __syncthreads();

        Frag16 af[4], bf[4];
        #pragma unroll
        for (int mt = 0; mt < 4; ++mt) {
            int row = 64 * wr + 16 * mt + lm;
            af[mt].q[0] = *(const uint4*)&As[row * 32 + 8 * hi];
            af[mt].q[1] = *(const uint4*)&As[row * 32 + 8 * hi + 16];
        }
        #pragma unroll
        for (int nt = 0; nt < 4; ++nt) {
            int col = 64 * wc + 16 * nt + lm;
            const uint4* p = (const uint4*)&Bs[col * 32 + 8 * hi];
            bf[nt].q[0] = p[0]; bf[nt].q[1] = p[2];
        }
        #pragma unroll
        for (int mt = 0; mt < 4; ++mt)
            #pragma unroll
            for (int nt = 0; nt < 4; ++nt)
                acc[mt][nt] = wmma_bf16(af[mt], bf[nt], acc[mt][nt]);
        __syncthreads();
    }

    // epilogue: one uniform branch, straight-line stores inside
    if (Cb) {
        #pragma unroll
        for (int nt = 0; nt < 4; ++nt) {
            int col = n0 + 64 * wc + 16 * nt + lm;
            float bval = bias[col];
            #pragma unroll
            for (int mt = 0; mt < 4; ++mt)
                #pragma unroll
                for (int j = 0; j < 8; ++j) {
                    int row = m0 + 64 * wr + 16 * mt + j + 8 * hi;
                    Cb[(size_t)row * Nn + col] = f2bf(acc[mt][nt][j] + bval);
                }
        }
    } else {
        #pragma unroll
        for (int nt = 0; nt < 4; ++nt) {
            int col = n0 + 64 * wc + 16 * nt + lm;
            float bval = bias[col];
            #pragma unroll
            for (int mt = 0; mt < 4; ++mt)
                #pragma unroll
                for (int j = 0; j < 8; ++j) {
                    int row = m0 + 64 * wr + 16 * mt + j + 8 * hi;
                    Cf[(size_t)row * Nn + col] = acc[mt][nt][j] + bval;
                }
        }
    }
}

// ---------------------------------------------------------------- flash attention
// q == k == qk_bf head-slice; out = softmax(q k^T / 8) v, bf16 out (B,N,C) layout
__global__ void __launch_bounds__(128)
k_flash_attn(const unsigned short* __restrict__ qk_bf,
             const unsigned short* __restrict__ v_bf,
             unsigned short* __restrict__ osp_bf)
{
    __shared__ __align__(16) unsigned short Kt[64 * 64];      // [k_token][d] row-major
    __shared__ __align__(16) unsigned short Vt[64 * 64];      // [d_col][k_row] transposed
    __shared__ __align__(16) unsigned short Pb[4 * 16 * 64];  // per-wave P tile [row][col]

    const int t = threadIdx.x;
    const int w = t >> 5, lane = t & 31;
    const int lm = lane & 15, hi = lane >> 4;
    const int rb = blockIdx.x * 64;            // row block (tokens)
    const int h  = blockIdx.y;
    const int b  = blockIdx.z;
    const float scale = 0.125f;                // 1/sqrt(64)

    const size_t headBase = ((size_t)b * N_) * C_ + (size_t)h * D_;

    // persistent q fragments: wave rows rb+16w .. +15, K = d (2 k-steps of 32)
    Frag16 qf[2];
    {
        const unsigned short* qp = qk_bf + headBase + (size_t)(rb + 16 * w + lm) * C_;
        #pragma unroll
        for (int s = 0; s < 2; ++s) {
            qf[s].q[0] = *(const uint4*)(qp + 32 * s + 8 * hi);
            qf[s].q[1] = *(const uint4*)(qp + 32 * s + 8 * hi + 16);
        }
    }

    v8f Oacc[4];
    #pragma unroll
    for (int nt = 0; nt < 4; ++nt)
        #pragma unroll
        for (int e = 0; e < 8; ++e) Oacc[nt][e] = 0.f;
    float mrow[8], lrow[8];
    #pragma unroll
    for (int j = 0; j < 8; ++j) { mrow[j] = -3.0e38f; lrow[j] = 0.f; }

    for (int cb = 0; cb < N_ / 64; ++cb) {
#if HAVE_TDM
        // K tile DMA: 64 token rows x 64 halves, pitch C_, row-major into Kt
        if (w == 0) {
            tdm_load_2d(lds_off32(Kt),
                        qk_bf + headBase + (size_t)(cb * 64) * C_,
                        (unsigned)C_, (unsigned)N_, 64u, 64u, (unsigned)C_);
        }
#else
        #pragma unroll
        for (int i = 0; i < 8; ++i) {
            int c  = t + 128 * i;
            int kr = c >> 4;
            int dc = c & 15;
            *(uint2*)&Kt[kr * 64 + 4 * dc] =
                *(const uint2*)(qk_bf + headBase + (size_t)(cb * 64 + kr) * C_ + 4 * dc);
        }
#endif
        // stage V tile transposed: Vt[d][krow]
        #pragma unroll
        for (int i = 0; i < 8; ++i) {
            int c  = t + 128 * i;              // 0..1023
            int kr = c >> 4;                   // 0..63
            int dc = c & 15;                   // chunk of 4 halves
            uint2 dv = *(const uint2*)(v_bf + headBase + (size_t)(cb * 64 + kr) * C_ + 4 * dc);
            int nb = 4 * dc;
            Vt[(nb + 0) * 64 + kr] = (unsigned short)(dv.x & 0xFFFF);
            Vt[(nb + 1) * 64 + kr] = (unsigned short)(dv.x >> 16);
            Vt[(nb + 2) * 64 + kr] = (unsigned short)(dv.y & 0xFFFF);
            Vt[(nb + 3) * 64 + kr] = (unsigned short)(dv.y >> 16);
        }
        // prefetch next block's V rows into cache while we compute
        if (cb + 1 < N_ / 64) {
            __builtin_prefetch(v_bf + headBase + (size_t)((cb + 1) * 64 + (t & 63)) * C_, 0, 0);
        }
#if HAVE_TDM
        if (w == 0) __builtin_amdgcn_s_wait_tensorcnt(0);
#endif
        __syncthreads();

        // S = q * k^T  (B fragment of column token j = row j of Kt, contiguous)
        v8f Sacc[4];
        #pragma unroll
        for (int nt = 0; nt < 4; ++nt)
            #pragma unroll
            for (int e = 0; e < 8; ++e) Sacc[nt][e] = 0.f;
        #pragma unroll
        for (int nt = 0; nt < 4; ++nt) {
            const unsigned short* kp = &Kt[(16 * nt + lm) * 64];
            #pragma unroll
            for (int s = 0; s < 2; ++s) {
                Frag16 kf;
                kf.q[0] = *(const uint4*)(kp + 32 * s + 8 * hi);
                kf.q[1] = *(const uint4*)(kp + 32 * s + 8 * hi + 16);
                Sacc[nt] = wmma_bf16(qf[s], kf, Sacc[nt]);
            }
        }

        // online softmax (row j -> VGPR slot j; reduce across 16-lane group)
        float fac[8], rs[8];
        #pragma unroll
        for (int j = 0; j < 8; ++j) {
            float mx = -3.0e38f;
            #pragma unroll
            for (int nt = 0; nt < 4; ++nt) mx = fmaxf(mx, Sacc[nt][j]);
            mx = fmaxf(mx, __shfl_xor(mx, 1, 16));
            mx = fmaxf(mx, __shfl_xor(mx, 2, 16));
            mx = fmaxf(mx, __shfl_xor(mx, 4, 16));
            mx = fmaxf(mx, __shfl_xor(mx, 8, 16));
            float mnew = fmaxf(mrow[j], mx * scale);
            float f = __expf(mrow[j] - mnew);
            mrow[j] = mnew;
            fac[j] = f;
            lrow[j] *= f;
            rs[j] = 0.f;
        }
        #pragma unroll
        for (int nt = 0; nt < 4; ++nt)
            #pragma unroll
            for (int j = 0; j < 8; ++j) Oacc[nt][j] *= fac[j];

        // P = exp(scale*S - m)  -> bf16 into wave-private LDS tile
        #pragma unroll
        for (int nt = 0; nt < 4; ++nt) {
            #pragma unroll
            for (int j = 0; j < 8; ++j) {
                float p = __expf(Sacc[nt][j] * scale - mrow[j]);
                rs[j] += p;
                Pb[w * 1024 + (j + 8 * hi) * 64 + 16 * nt + lm] = f2bf(p);
            }
        }
        #pragma unroll
        for (int j = 0; j < 8; ++j) {
            float s = rs[j];
            s += __shfl_xor(s, 1, 16);
            s += __shfl_xor(s, 2, 16);
            s += __shfl_xor(s, 4, 16);
            s += __shfl_xor(s, 8, 16);
            lrow[j] += s;
        }
        __syncthreads();

        // O += P * V
        #pragma unroll
        for (int s = 0; s < 2; ++s) {
            Frag16 pf;
            const unsigned short* pp = &Pb[w * 1024 + lm * 64 + 32 * s + 8 * hi];
            pf.q[0] = *(const uint4*)pp;
            pf.q[1] = *(const uint4*)(pp + 16);
            #pragma unroll
            for (int nt = 0; nt < 4; ++nt) {
                Frag16 vf;
                const unsigned short* vp = &Vt[(16 * nt + lm) * 64 + 32 * s + 8 * hi];
                vf.q[0] = *(const uint4*)vp;
                vf.q[1] = *(const uint4*)(vp + 16);
                Oacc[nt] = wmma_bf16(pf, vf, Oacc[nt]);
            }
        }
        __syncthreads();
    }

    // epilogue: divide by row sum, write bf16 into (B,N,C) layout
    #pragma unroll
    for (int nt = 0; nt < 4; ++nt) {
        #pragma unroll
        for (int j = 0; j < 8; ++j) {
            int n = rb + 16 * w + j + 8 * hi;
            int d = 16 * nt + lm;
            osp_bf[headBase + (size_t)n * C_ + d] = f2bf(Oacc[nt][j] / lrow[j]);
        }
    }
}

// ---------------------------------------------------------------- channel branch
// one wave per (b,n): kp = kt@WE+bE, vp = vt@WE+bE,
// attn = softmax_d(kp^T kt), x_CA = vp @ attn   (r=21, h=16, d=64)
__global__ void __launch_bounds__(32)
k_channel(const unsigned short* __restrict__ qk_bf,
          const unsigned short* __restrict__ v_bf,
          const float* __restrict__ WE, const float* __restrict__ bE,
          unsigned short* __restrict__ xca_bf)
{
    __shared__ float ktL[H_ * D_];
    __shared__ float vtL[H_ * D_];
    __shared__ float kpL[H_ * R_];
    __shared__ float vpL[H_ * R_];

    const int t = threadIdx.x;          // 0..31
    const size_t base = (size_t)blockIdx.x * C_;

    for (int i = t; i < H_ * D_; i += 32) {
        ktL[i] = bf2f(qk_bf[base + i]);
        vtL[i] = bf2f(v_bf[base + i]);
    }
    __syncthreads();

    for (int idx = t; idx < H_ * R_; idx += 32) {
        int hh = idx / R_, rr = idx % R_;
        float a0 = bE[rr], a1 = bE[rr];
        for (int dd = 0; dd < D_; ++dd) {
            float wv_ = WE[dd * R_ + rr];
            a0 += ktL[hh * D_ + dd] * wv_;
            a1 += vtL[hh * D_ + dd] * wv_;
        }
        kpL[idx] = a0;
        vpL[idx] = a1;
    }
    __syncthreads();

    // thread t owns d columns t and t+32
    float p0[R_], p1[R_];
    #pragma unroll 1
    for (int rr = 0; rr < R_; ++rr) {
        float s0 = 0.f, s1 = 0.f;
        for (int hh = 0; hh < H_; ++hh) {
            float kpv = kpL[hh * R_ + rr];
            s0 += kpv * ktL[hh * D_ + t];
            s1 += kpv * ktL[hh * D_ + t + 32];
        }
        float mx = fmaxf(s0, s1);
        mx = fmaxf(mx, __shfl_xor(mx, 1, 32));
        mx = fmaxf(mx, __shfl_xor(mx, 2, 32));
        mx = fmaxf(mx, __shfl_xor(mx, 4, 32));
        mx = fmaxf(mx, __shfl_xor(mx, 8, 32));
        mx = fmaxf(mx, __shfl_xor(mx, 16, 32));
        float e0 = __expf(s0 - mx), e1 = __expf(s1 - mx);
        float sm = e0 + e1;
        sm += __shfl_xor(sm, 1, 32);
        sm += __shfl_xor(sm, 2, 32);
        sm += __shfl_xor(sm, 4, 32);
        sm += __shfl_xor(sm, 8, 32);
        sm += __shfl_xor(sm, 16, 32);
        float inv = 1.f / sm;
        p0[rr] = e0 * inv;
        p1[rr] = e1 * inv;
    }

    for (int hh = 0; hh < H_; ++hh) {
        float a0 = 0.f, a1 = 0.f;
        for (int rr = 0; rr < R_; ++rr) {
            float vv = vpL[hh * R_ + rr];
            a0 += vv * p0[rr];
            a1 += vv * p1[rr];
        }
        xca_bf[base + hh * D_ + t]      = f2bf(a0);
        xca_bf[base + hh * D_ + t + 32] = f2bf(a1);
    }
}

// ---------------------------------------------------------------- launch
extern "C" void kernel_launch(void* const* d_in, const int* in_sizes, int n_in,
                              void* d_out, int out_size, void* d_ws, size_t ws_size,
                              hipStream_t stream) {
    const float* x    = (const float*)d_in[0];
    const float* Wqk  = (const float*)d_in[1];
    const float* bqk  = (const float*)d_in[2];
    const float* Wv   = (const float*)d_in[3];
    const float* bv   = (const float*)d_in[4];
    const float* Wout = (const float*)d_in[5];
    const float* bout = (const float*)d_in[6];
    const float* WE   = (const float*)d_in[7];
    const float* bE   = (const float*)d_in[8];
    float* out = (float*)d_out;

    char* ws = (char*)d_ws;
    size_t off = 0;
    auto alloc = [&](size_t bytes) -> char* {
        char* p = ws + off;
        off += (bytes + 255) & ~(size_t)255;
        return p;
    };
    const size_t NE = (size_t)MTOK * C_;   // 4M elements
    unsigned short* x_bf    = (unsigned short*)alloc(NE * 2);
    unsigned short* wqk_bf  = (unsigned short*)alloc((size_t)C_ * C_ * 2);
    unsigned short* wv_bf   = (unsigned short*)alloc((size_t)C_ * C_ * 2);
    unsigned short* wout_bf = (unsigned short*)alloc((size_t)C_ * C_ * 2);
    unsigned short* qk_bf   = (unsigned short*)alloc(NE * 2);
    unsigned short* v_bf    = (unsigned short*)alloc(NE * 2);
    unsigned short* osp_bf  = (unsigned short*)alloc(NE * 2);
    unsigned short* xca_bf  = (unsigned short*)alloc(NE * 2);
    unsigned short* sum_bf  = (unsigned short*)alloc(NE * 2);

    k_f32_to_bf16<<<512, 256, 0, stream>>>(x, x_bf, (int)NE);
    k_f32_to_bf16<<<256, 256, 0, stream>>>(Wqk, wqk_bf, C_ * C_);
    k_f32_to_bf16<<<256, 256, 0, stream>>>(Wv, wv_bf, C_ * C_);
    k_f32_to_bf16<<<256, 256, 0, stream>>>(Wout, wout_bf, C_ * C_);

    dim3 ggrid(MTOK / 128, C_ / 128);
    k_gemm_bf16<<<ggrid, 128, 0, stream>>>(x_bf, wqk_bf, bqk, nullptr, qk_bf, MTOK, C_, C_);
    k_gemm_bf16<<<ggrid, 128, 0, stream>>>(x_bf, wv_bf,  bv,  nullptr, v_bf,  MTOK, C_, C_);

    dim3 agrid(N_ / 64, H_, B_);
    k_flash_attn<<<agrid, 128, 0, stream>>>(qk_bf, v_bf, osp_bf);

    k_channel<<<MTOK, 32, 0, stream>>>(qk_bf, v_bf, WE, bE, xca_bf);

    k_halfsum_bf16<<<512, 256, 0, stream>>>(osp_bf, xca_bf, sum_bf, (int)NE);

    // 0.5*(out + x_CA) = (0.5*(attn_sp + x_CA)) @ Wout + bout
    k_gemm_bf16<<<ggrid, 128, 0, stream>>>(sum_bf, wout_bf, bout, out, nullptr, MTOK, C_, C_);
}
